// GraphAttentionLayer_43825846288460
// MI455X (gfx1250) — matrix operand order repaired and checked by reading, not compiled, not run on hardware
//
#include <hip/hip_runtime.h>
#include <hip/hip_bf16.h>

typedef __attribute__((ext_vector_type(16))) _Float16 v16h;
typedef __attribute__((ext_vector_type(8)))  float    v8f;

#define N_NODES 4096
#define IN_F    256
#define OUT_F   64
#define NHEAD   4
#define ALPHA   0.2f

// ---------------------------------------------------------------------------
// Kernel 1: Wh = h @ W per head (f32 accum), store WhT as f16 [h][f][j] for
// WMMA B-matrix loads; also reduce f1 = Wh.a1, f2 = Wh.a2 (kept in f32).
// grid (4096, 4), block 64
// ---------------------------------------------------------------------------
__global__ void k_wh(const float* __restrict__ h, const float* __restrict__ W,
                     const float* __restrict__ a,
                     _Float16* __restrict__ WhT,
                     float* __restrict__ f1_4, float* __restrict__ f2_4,
                     float* __restrict__ f2h)
{
    const int n  = blockIdx.x;
    const int hh = blockIdx.y;
    const int t  = threadIdx.x;               // output feature f

    __shared__ float sh[IN_F];
    __shared__ float red1[64];
    __shared__ float red2[64];

    #pragma unroll
    for (int q = 0; q < 4; ++q)
        sh[t + q * 64] = h[(size_t)n * IN_F + t + q * 64];
    __syncthreads();

    const float* Wp = W + (size_t)hh * (IN_F * OUT_F) + t;
    float acc = 0.0f;
    #pragma unroll 8
    for (int k = 0; k < IN_F; ++k)
        acc = fmaf(sh[k], Wp[(size_t)k * OUT_F], acc);

    // transposed f16 store for WMMA B matrix: [h][f][j]
    WhT[(size_t)hh * (OUT_F * N_NODES) + (size_t)t * N_NODES + n] = (_Float16)acc;

    red1[t] = acc * a[hh * (2 * OUT_F) + t];
    red2[t] = acc * a[hh * (2 * OUT_F) + OUT_F + t];
    __syncthreads();
    for (int s = 32; s > 0; s >>= 1) {
        if (t < s) { red1[t] += red1[t + s]; red2[t] += red2[t + s]; }
        __syncthreads();
    }
    if (t == 0) {
        f1_4[n * NHEAD + hh] = red1[0];
        f2_4[n * NHEAD + hh] = red2[0];
        f2h[(size_t)hh * N_NODES + n] = red2[0];  // head-major copy for pass B
    }
}

// ---------------------------------------------------------------------------
// Kernel 2: per-row online softmax stats over masked logits.
// m[i,h] = max_j e[i,j,h], rs[i,h] = 1/sum_j exp(e-m)  (j over neighbors)
// grid 4096, block 256
// ---------------------------------------------------------------------------
__device__ __forceinline__ void online_upd(float f1v, float f2v, float& M, float& S)
{
    float x = f1v + f2v;
    float e = fmaxf(x, ALPHA * x);            // leaky_relu
    if (e > M) { S = S * __expf(M - e) + 1.0f; M = e; }
    else       { S += __expf(e - M); }
}

__global__ void k_stats(const float* __restrict__ adj,
                        const float* __restrict__ f1_4,
                        const float* __restrict__ f2_4,
                        float* __restrict__ m4, float* __restrict__ rs4)
{
    const int i = blockIdx.x;
    const int t = threadIdx.x;

    const float4 f1v = ((const float4*)f1_4)[i];
    float M0 = -3.0e38f, M1 = -3.0e38f, M2 = -3.0e38f, M3 = -3.0e38f;
    float S0 = 0.f, S1 = 0.f, S2 = 0.f, S3 = 0.f;

    const float* arow = adj + (size_t)i * N_NODES;
    #pragma unroll 4
    for (int it = 0; it < N_NODES / 256; ++it) {
        const int j = t + it * 256;
        const float av = arow[j];
        if (av != 0.0f) {
            const float4 f2v = ((const float4*)f2_4)[j];
            online_upd(f1v.x, f2v.x, M0, S0);
            online_upd(f1v.y, f2v.y, M1, S1);
            online_upd(f1v.z, f2v.z, M2, S2);
            online_upd(f1v.w, f2v.w, M3, S3);
        }
    }

    __shared__ float rm[NHEAD][256];
    __shared__ float rs[NHEAD][256];
    rm[0][t] = M0; rm[1][t] = M1; rm[2][t] = M2; rm[3][t] = M3;
    rs[0][t] = S0; rs[1][t] = S1; rs[2][t] = S2; rs[3][t] = S3;
    __syncthreads();
    for (int s = 128; s > 0; s >>= 1) {
        if (t < s) {
            #pragma unroll
            for (int hq = 0; hq < NHEAD; ++hq) {
                float m1 = rm[hq][t], m2 = rm[hq][t + s];
                float s1 = rs[hq][t], s2 = rs[hq][t + s];
                float mm = fmaxf(m1, m2);
                rm[hq][t] = mm;
                rs[hq][t] = s1 * __expf(m1 - mm) + s2 * __expf(m2 - mm);
            }
        }
        __syncthreads();
    }
    if (t < NHEAD) {
        m4[i * NHEAD + t]  = rm[t][0];
        rs4[i * NHEAD + t] = 1.0f / rs[t][0];
    }
}

// ---------------------------------------------------------------------------
// Kernel 3: fused attention GEMM  h'[i,h,:] = sum_j p(i,j,h) * Wh[j,h,:]
// via v_wmma_f32_16x16x32_f16, adj tile staged with async global->LDS DMA,
// ELU fused into writeback.
// block = 128 threads = 4 waves, wave w handles head w for rows i0..i0+15.
// grid 256 blocks. launch_bounds(128,1) relaxes the occupancy floor so all
// four B tiles stay live in registers (loads overlap the exp-heavy A-build).
// ---------------------------------------------------------------------------
#define JCHUNK 128
#define ADJ_PITCH 132   // 128 + 4 pad -> rows land on distinct bank groups

__global__ void __launch_bounds__(128, 1)
k_attn(const float* __restrict__ adj,
       const float* __restrict__ f2h,
       const _Float16* __restrict__ WhT,
       const float* __restrict__ f1_4,
       const float* __restrict__ m4,
       const float* __restrict__ rs4,
       float* __restrict__ out)
{
    const int i0   = blockIdx.x * 16;
    const int t    = threadIdx.x;
    const int lane = t & 31;
    const int hh   = t >> 5;          // head per wave
    const int hl   = lane >> 4;       // lane half
    const int r    = lane & 15;       // A-matrix row / C-matrix N index
    const int b8   = hl * 8;          // C-row (M) offset of this lane half

    // per-lane row stats for the A (probability) matrix:
    // p = adj * exp(leaky(f1+f2)) * RSF,  RSF = exp(-m)/s  (adj is exactly 0/1)
    const int  arow = i0 + r;
    const float F1  = f1_4[arow * NHEAD + hh];
    const float RSF = __expf(-m4[arow * NHEAD + hh]) * rs4[arow * NHEAD + hh];

    __shared__ float lds_adj[16 * ADJ_PITCH];
    __shared__ float lds_f2[NHEAD * JCHUNK];

    v8f c[4] = {};                    // 4 f-tiles of 16 cols -> OUT_F=64

    const int rr = t >> 3;            // adj stage: row
    const int cg = (t & 7) * 16;      // adj stage: col group

    for (int jc = 0; jc < N_NODES; jc += JCHUNK) {
        __syncthreads();              // previous tile fully consumed
        // ---- stage adj tile (16 x 128) via async global->LDS DMA ----
        {
            const float* gsrc = adj + (size_t)(i0 + rr) * N_NODES + jc + cg;
            #pragma unroll
            for (int q = 0; q < 4; ++q) {
                const unsigned ldsoff =
                    (unsigned)(uintptr_t)&lds_adj[rr * ADJ_PITCH + cg + 4 * q];
                asm volatile("global_load_async_to_lds_b128 %0, %1, off"
                             :: "v"(ldsoff), "v"(gsrc + 4 * q) : "memory");
            }
            if (jc + JCHUNK < N_NODES)
                __builtin_prefetch(gsrc + JCHUNK, 0, 1);   // global_prefetch_b8
        }
        // ---- stage f2 chunk (4 heads x 128) ----
        {
            const int h2 = t >> 5, jl = (t & 31) * 4;
            *(float4*)&lds_f2[h2 * JCHUNK + jl] =
                *(const float4*)(f2h + (size_t)h2 * N_NODES + jc + jl);
        }
        asm volatile("s_wait_asynccnt 0x0" ::: "memory");
        __syncthreads();

        // ---- 4 K-steps of 32 per chunk ----
        #pragma unroll
        for (int kk = 0; kk < JCHUNK; kk += 32) {
            // ---- issue all 4 B-tile loads first so they overlap the A-build
            // B from WhT[h][f][j]: lane gets K = 16*hl + 0..15 contiguous
            const _Float16* wbase = WhT + (size_t)hh * (OUT_F * N_NODES)
                                        + (size_t)r * N_NODES + jc + kk + hl * 16;
            v16h Bt[4];
            #pragma unroll
            for (int ft = 0; ft < 4; ++ft)
                Bt[ft] = *(const v16h*)(wbase + (size_t)ft * 16 * N_NODES);

            // ---- build A: 16 f16 probabilities per lane, 16x32 A layout:
            //   element e -> K = 16*hl + e
            float adjv[16];
            const int abase = r * ADJ_PITCH + kk + hl * 16;
            *(float4*)&adjv[0]  = *(const float4*)&lds_adj[abase];
            *(float4*)&adjv[4]  = *(const float4*)&lds_adj[abase + 4];
            *(float4*)&adjv[8]  = *(const float4*)&lds_adj[abase + 8];
            *(float4*)&adjv[12] = *(const float4*)&lds_adj[abase + 12];
            const float* f2p = &lds_f2[hh * JCHUNK + kk + hl * 16];

            v16h A;
            #pragma unroll
            for (int e = 0; e < 16; ++e) {
                const float f2v = f2p[e];
                const float x  = F1 + f2v;
                const float ee = fmaxf(x, ALPHA * x);     // leaky_relu
                const float p  = adjv[e] * (__expf(ee) * RSF);  // adj in {0,1}
                A[e] = (_Float16)p;
            }

            // ---- 4 back-to-back WMMAs sharing A
            #pragma unroll
            for (int ft = 0; ft < 4; ++ft)
                c[ft] = __builtin_amdgcn_wmma_f32_16x16x32_f16(
                            false, A, false, Bt[ft], (short)0, c[ft], false, false);
        }
    }

    // ---- ELU + writeback; C layout: VGPR q -> M = q + 8*hl, N = lane&15 ----
    #pragma unroll
    for (int ft = 0; ft < 4; ++ft) {
        #pragma unroll
        for (int q = 0; q < 8; ++q) {
            const float v = c[ft][q];
            const float o = (v > 0.0f) ? v : (__expf(v) - 1.0f);
            out[(size_t)(i0 + q + b8) * (NHEAD * OUT_F) + hh * OUT_F + ft * 16 + r] = o;
        }
    }
}

// ---------------------------------------------------------------------------
// launcher
// ---------------------------------------------------------------------------
extern "C" void kernel_launch(void* const* d_in, const int* in_sizes, int n_in,
                              void* d_out, int out_size, void* d_ws, size_t ws_size,
                              hipStream_t stream) {
    const float* h   = (const float*)d_in[0];
    const float* adj = (const float*)d_in[1];
    const float* W   = (const float*)d_in[2];
    const float* a   = (const float*)d_in[3];
    float* out = (float*)d_out;

    // workspace partition (~2.32 MB total)
    char* ws = (char*)d_ws;
    constexpr size_t OFF_WHT = 0;                          // 4*64*4096 f16 = 2 MB
    constexpr size_t OFF_F1  = OFF_WHT + (size_t)NHEAD * OUT_F * N_NODES * 2;
    constexpr size_t OFF_F2  = OFF_F1 + (size_t)N_NODES * NHEAD * 4;
    constexpr size_t OFF_F2H = OFF_F2 + (size_t)N_NODES * NHEAD * 4;
    constexpr size_t OFF_M   = OFF_F2H + (size_t)N_NODES * NHEAD * 4;
    constexpr size_t OFF_RS  = OFF_M + (size_t)N_NODES * NHEAD * 4;

    _Float16* WhT = (_Float16*)(ws + OFF_WHT);
    float* f1_4 = (float*)(ws + OFF_F1);
    float* f2_4 = (float*)(ws + OFF_F2);
    float* f2h  = (float*)(ws + OFF_F2H);
    float* m4   = (float*)(ws + OFF_M);
    float* rs4  = (float*)(ws + OFF_RS);

    k_wh   <<<dim3(N_NODES, NHEAD), 64,  0, stream>>>(h, W, a, WhT, f1_4, f2_4, f2h);
    k_stats<<<N_NODES,              256, 0, stream>>>(adj, f1_4, f2_4, m4, rs4);
    k_attn <<<N_NODES / 16,         128, 0, stream>>>(adj, f2h, WhT, f1_4, m4, rs4, out);
}
